// SparseAttention_63196148793906
// MI455X (gfx1250) — matrix operand order repaired and checked by reading, not compiled, not run on hardware
//
#include <hip/hip_runtime.h>
#include <math.h>

// ---------------------------------------------------------------------------
// CDNA5 (gfx1250) sparse local+global attention, bf16 WMMA + f32 accumulate.
// Pipeline: f32->bf16 cast; Q/K/V projections (TDM-fed LDS tiled WMMA GEMM,
// V stored transposed); flash-style sparse attention (one wave per 16-row
// query tile); output projection GEMM -> f32.
// ---------------------------------------------------------------------------

typedef __attribute__((ext_vector_type(16))) __bf16 v16bf;
typedef __attribute__((ext_vector_type(8)))  __bf16 v8bf;
typedef __attribute__((ext_vector_type(8)))  float  v8f;
typedef __attribute__((ext_vector_type(4)))  unsigned int v4u;
typedef __attribute__((ext_vector_type(8)))  int v8i;
typedef __attribute__((ext_vector_type(4)))  int v4i;

#define S_LEN  4096
#define HDIM   1024
#define NHEADS 16
#define HEADD  64
#define HALFW  128
#define NGLOB  64

__device__ __forceinline__ v8bf ld8(const __bf16* p) { return *(const v8bf*)p; }

__device__ __forceinline__ v16bf comb(v8bf lo, v8bf hi) {
  v16bf r;
#pragma unroll
  for (int i = 0; i < 8; ++i) { r[i] = lo[i]; r[i + 8] = hi[i]; }
  return r;
}

__device__ __forceinline__ v8f wmma_bf16(v16bf a, v16bf b, v8f c) {
  // D = A(16x32) * B(32x16) + C, f32 accumulate
  return __builtin_amdgcn_wmma_f32_16x16x32_bf16(false, a, false, b,
                                                 (short)0, c, false, false);
}

__device__ __forceinline__ float redmax16(float t) {
#pragma unroll
  for (int off = 1; off < 16; off <<= 1) t = fmaxf(t, __shfl_xor(t, off, 32));
  return t;
}
__device__ __forceinline__ float redsum16(float t) {
#pragma unroll
  for (int off = 1; off < 16; off <<= 1) t += __shfl_xor(t, off, 32);
  return t;
}

// ---------------------------------------------------------------------------
// TDM: async 2D tile load (32 elem wide x 128 rows, bf16) global -> LDS with
// LDS padding of 4 dwords every 16 dwords (row stride 80B -> conflict-free
// b128 fragment reads). D# per CDNA5 ISA.
// 6-arg builtin form (clang-23 / therock headers).
// ---------------------------------------------------------------------------
__device__ __forceinline__ void tdm_load_tile(unsigned lds_off, const __bf16* g,
                                              unsigned tdim0, unsigned tdim1) {
  unsigned long long ga = (unsigned long long)(size_t)g;
  v4u g0 = { 1u,                                   // count=1 (valid descriptor)
             lds_off,                              // lds_addr (bytes)
             (unsigned)(ga & 0xFFFFFFFFu),         // global_addr[31:0]
             (unsigned)((ga >> 32) & 0x1FFFFFFu) | (2u << 30) }; // [56:32]|type=2
  v8i g1;
  // data_size=1(2B) | pad_enable | pad_interval=3(16dw) | pad_amount=3(4dw)
  g1[0] = (int)((1u << 16) | (1u << 20) | (3u << 22) | (3u << 25));
  g1[1] = (int)((tdim0 & 0xFFFFu) << 16);                       // tensor_dim0 lo
  g1[2] = (int)(((tdim0 >> 16) & 0xFFFFu) | ((tdim1 & 0xFFFFu) << 16));
  g1[3] = (int)(((tdim1 >> 16) & 0xFFFFu) | (32u << 16));       // tile_dim0=32
  g1[4] = 128;                                                  // tile_dim1=128
  g1[5] = (int)tdim0;                                           // dim0 stride
  g1[6] = 0; g1[7] = 0;
  v4i z4 = { 0, 0, 0, 0 };
  v8i z8 = { 0, 0, 0, 0, 0, 0, 0, 0 };
  __builtin_amdgcn_tensor_load_to_lds(g0, g1, z4, z4, z8, 0);
}

// ---------------------------------------------------------------------------
// f32 -> bf16 cast
// ---------------------------------------------------------------------------
__global__ __launch_bounds__(256) void cvt_bf16(const float* __restrict__ src,
                                                __bf16* __restrict__ dst, int n) {
  int i = blockIdx.x * 256 + threadIdx.x;
  int stride = gridDim.x * 256;
  for (; i < n; i += stride) dst[i] = (__bf16)src[i];
}

// ---------------------------------------------------------------------------
// C[4096x1024] = A[4096x1024] * B[1024x1024]^T  (Linear: both K-contiguous)
// Block: 256 thr = 8 waves; block tile 128x128; wave tile 32x64 (8 WMMA/step)
// OUTMODE 0: bf16 row-major * scale; 1: bf16 transposed (Vt); 2: f32 row-major
// ---------------------------------------------------------------------------
template <int OUTMODE>
__global__ __launch_bounds__(256) void gemm128(const __bf16* __restrict__ A,
                                               const __bf16* __restrict__ B,
                                               void* __restrict__ Cp,
                                               float scale) {
  constexpr int M = 4096, N = 1024, K = 1024;
  constexpr int LDT = 40;                  // padded LDS row stride (elements)
  __shared__ __bf16 tA[2][128 * LDT];
  __shared__ __bf16 tB[2][128 * LDT];

  const int tid  = threadIdx.x;
  const int wave = tid >> 5, lane = tid & 31;
  const int lr = lane & 15, lh = lane >> 4;
  const int wm = wave >> 1, wn = wave & 1;
  const int m0 = blockIdx.y * 128, n0 = blockIdx.x * 128;

  v8f c[2][4];
#pragma unroll
  for (int i = 0; i < 2; ++i)
#pragma unroll
    for (int j = 0; j < 4; ++j)
#pragma unroll
      for (int r = 0; r < 8; ++r) c[i][j][r] = 0.0f;

  constexpr int T = K / 32;
  if (wave == 0) {
    tdm_load_tile((unsigned)(size_t)&tA[0][0], A + (size_t)m0 * K, K, M);
    tdm_load_tile((unsigned)(size_t)&tB[0][0], B + (size_t)n0 * K, K, N);
  }

  for (int t = 0; t < T; ++t) {
    const int cur = t & 1, nxt = cur ^ 1;
    if (wave == 0) {
      if (t + 1 < T) {
        const int k1 = (t + 1) * 32;
        tdm_load_tile((unsigned)(size_t)&tA[nxt][0], A + (size_t)m0 * K + k1, K, M);
        tdm_load_tile((unsigned)(size_t)&tB[nxt][0], B + (size_t)n0 * K + k1, K, N);
        __builtin_amdgcn_s_wait_tensorcnt(2);   // current pair complete
      } else {
        __builtin_amdgcn_s_wait_tensorcnt(0);
      }
    }
    __syncthreads();

    v16bf af[2], bfr[4];
    {
      const __bf16* p = &tA[cur][(wm * 32 + lr) * LDT + lh * 16];
      af[0] = comb(ld8(p), ld8(p + 8));           // K 0..15 / 16..31 per half
      p += 16 * LDT;
      af[1] = comb(ld8(p), ld8(p + 8));
    }
    {
      const __bf16* p = &tB[cur][(wn * 64 + lr) * LDT + lh * 16];
      bfr[0] = comb(ld8(p), ld8(p + 8));          // K 0..15 / 16..31 per half
      p += 16 * LDT; bfr[1] = comb(ld8(p), ld8(p + 8));
      p += 16 * LDT; bfr[2] = comb(ld8(p), ld8(p + 8));
      p += 16 * LDT; bfr[3] = comb(ld8(p), ld8(p + 8));
    }
#pragma unroll
    for (int mi = 0; mi < 2; ++mi)
#pragma unroll
      for (int ni = 0; ni < 4; ++ni)
        c[mi][ni] = wmma_bf16(af[mi], bfr[ni], c[mi][ni]);
    __syncthreads();
  }

#pragma unroll
  for (int mi = 0; mi < 2; ++mi)
#pragma unroll
    for (int ni = 0; ni < 4; ++ni)
#pragma unroll
      for (int r = 0; r < 8; ++r) {
        const int gm = m0 + wm * 32 + mi * 16 + r + lh * 8;   // C: M=r+8*half
        const int gn = n0 + wn * 64 + ni * 16 + lr;           // C: N=lane&15
        const float v = c[mi][ni][r] * scale;
        if (OUTMODE == 0)      ((__bf16*)Cp)[(size_t)gm * N + gn] = (__bf16)v;
        else if (OUTMODE == 1) ((__bf16*)Cp)[(size_t)gn * M + gm] = (__bf16)v;
        else                   ((float*)Cp)[(size_t)gm * N + gn] = v;
      }
}

// ---------------------------------------------------------------------------
// Sparse flash attention. One wave per (head, 16-row query tile).
// Mask: |i-j|<=128 || j<64 || i<64. Scale folded into Q.
// Q,K: bf16 [4096][1024] row-major; Vt: bf16 [1024][4096]; O: bf16 [4096][1024]
// ---------------------------------------------------------------------------
__global__ __launch_bounds__(32) void attn_sparse(const __bf16* __restrict__ Q,
                                                  const __bf16* __restrict__ Km,
                                                  const __bf16* __restrict__ Vt,
                                                  __bf16* __restrict__ O) {
  constexpr int LDP = 40;
  __shared__ __bf16 P[16 * LDP];
  const int lane = threadIdx.x & 31;
  const int lr = lane & 15, lh = lane >> 4;
  const int h  = blockIdx.y;
  const int i0 = blockIdx.x * 16;
  const float NEG_INF = -__builtin_inff();

  v16bf q0, q1;
  {
    const __bf16* qp = Q + (size_t)(i0 + lr) * HDIM + h * HEADD + lh * 16;
    q0 = comb(ld8(qp),      ld8(qp + 8));    // head-dim 0..31
    q1 = comb(ld8(qp + 32), ld8(qp + 40));   // head-dim 32..63
  }

  v8f o0, o1, o2, o3;
  float mrow[8], lrow[8];
#pragma unroll
  for (int r = 0; r < 8; ++r) {
    o0[r] = 0.0f; o1[r] = 0.0f; o2[r] = 0.0f; o3[r] = 0.0f;
    mrow[r] = NEG_INF; lrow[r] = 0.0f;
  }

  for (int j0 = 0; j0 < S_LEN; j0 += 32) {
    const bool any = (i0 < NGLOB) || (j0 < NGLOB) ||
                     ((j0 + 31 >= i0 - HALFW) && (j0 <= i0 + 15 + HALFW));
    if (!any) continue;

    // K^T fragments: lane lr <-> key column, per-lane head-dim contiguous
    const __bf16* kp = Km + (size_t)(j0 + lr) * HDIM + h * HEADD + lh * 16;
    const __bf16* kq = kp + (size_t)16 * HDIM;
    v16bf kb00 = comb(ld8(kp),      ld8(kp + 8));
    v16bf kb01 = comb(ld8(kp + 32), ld8(kp + 40));
    v16bf kb10 = comb(ld8(kq),      ld8(kq + 8));
    v16bf kb11 = comb(ld8(kq + 32), ld8(kq + 40));

    v8f s0, s1;
#pragma unroll
    for (int r = 0; r < 8; ++r) { s0[r] = 0.0f; s1[r] = 0.0f; }
    s0 = wmma_bf16(q0, kb00, s0); s0 = wmma_bf16(q1, kb01, s0);
    s1 = wmma_bf16(q0, kb10, s1); s1 = wmma_bf16(q1, kb11, s1);

    const int jA = j0 + lr, jB = j0 + 16 + lr;
    float alpha[8];
#pragma unroll
    for (int r = 0; r < 8; ++r) {
      const int i = i0 + r + lh * 8;
      const bool gi = (i < NGLOB);
      const float x0 = (gi || jA < NGLOB || (jA >= i - HALFW && jA <= i + HALFW))
                           ? s0[r] : NEG_INF;
      const float x1 = (gi || jB < NGLOB || (jB >= i - HALFW && jB <= i + HALFW))
                           ? s1[r] : NEG_INF;
      const float t  = redmax16(fmaxf(x0, x1));
      const float mn = fmaxf(fmaxf(mrow[r], t), -1e30f);
      const float al = __expf(mrow[r] - mn);
      const float p0 = __expf(x0 - mn);
      const float p1 = __expf(x1 - mn);
      lrow[r] = lrow[r] * al + redsum16(p0 + p1);
      mrow[r] = mn;
      alpha[r] = al;
      const int row = r + lh * 8;
      P[row * LDP + lr]      = (__bf16)p0;
      P[row * LDP + 16 + lr] = (__bf16)p1;
    }
#pragma unroll
    for (int r = 0; r < 8; ++r) {
      o0[r] *= alpha[r]; o1[r] *= alpha[r];
      o2[r] *= alpha[r]; o3[r] *= alpha[r];
    }
    __syncthreads();  // single-wave WG: emits waits, barrier is free

    const __bf16* pp = &P[lr * LDP + lh * 16];
    v16bf pa = comb(ld8(pp), ld8(pp + 8));    // P as A fragment (16q x 32keys)

    const __bf16* vp = Vt + (size_t)(h * HEADD + lr) * S_LEN + j0 + lh * 16;
    v16bf vb0 = comb(ld8(vp),                ld8(vp + 8));
    v16bf vb1 = comb(ld8(vp + 16 * S_LEN),   ld8(vp + 16 * S_LEN + 8));
    v16bf vb2 = comb(ld8(vp + 32 * S_LEN),   ld8(vp + 32 * S_LEN + 8));
    v16bf vb3 = comb(ld8(vp + 48 * S_LEN),   ld8(vp + 48 * S_LEN + 8));
    o0 = wmma_bf16(pa, vb0, o0);
    o1 = wmma_bf16(pa, vb1, o1);
    o2 = wmma_bf16(pa, vb2, o2);
    o3 = wmma_bf16(pa, vb3, o3);
    __syncthreads();
  }

#pragma unroll
  for (int r = 0; r < 8; ++r) {
    const float inv = 1.0f / lrow[r];
    const size_t ro = (size_t)(i0 + r + lh * 8) * HDIM + h * HEADD + lr;
    O[ro]      = (__bf16)(o0[r] * inv);
    O[ro + 16] = (__bf16)(o1[r] * inv);
    O[ro + 32] = (__bf16)(o2[r] * inv);
    O[ro + 48] = (__bf16)(o3[r] * inv);
  }
}

// ---------------------------------------------------------------------------
// Host launcher. ws layout (needs 48 MB):
//  [0,8)MB Xb | [8,10) Wqb | [10,12) Wkb | [12,14) Wvb | [14,16) Wob
//  [16,24) Q | [24,32) K | [32,40) Vt | [40,48) O   (all bf16)
// ---------------------------------------------------------------------------
extern "C" void kernel_launch(void* const* d_in, const int* in_sizes, int n_in,
                              void* d_out, int out_size, void* d_ws, size_t ws_size,
                              hipStream_t stream) {
  (void)in_sizes; (void)n_in; (void)out_size; (void)ws_size;
  const float* X  = (const float*)d_in[0];
  const float* Wq = (const float*)d_in[1];
  const float* Wk = (const float*)d_in[2];
  const float* Wv = (const float*)d_in[3];
  const float* Wo = (const float*)d_in[4];

  char* ws = (char*)d_ws;
  const size_t MB = 1024ull * 1024ull;
  __bf16* Xb  = (__bf16*)(ws + 0 * MB);
  __bf16* Wqb = (__bf16*)(ws + 8 * MB);
  __bf16* Wkb = (__bf16*)(ws + 10 * MB);
  __bf16* Wvb = (__bf16*)(ws + 12 * MB);
  __bf16* Wob = (__bf16*)(ws + 14 * MB);
  __bf16* Qb  = (__bf16*)(ws + 16 * MB);
  __bf16* Kb  = (__bf16*)(ws + 24 * MB);
  __bf16* Vtb = (__bf16*)(ws + 32 * MB);
  __bf16* Ob  = (__bf16*)(ws + 40 * MB);

  cvt_bf16<<<1024, 256, 0, stream>>>(X,  Xb,  S_LEN * HDIM);
  cvt_bf16<<<512,  256, 0, stream>>>(Wq, Wqb, HDIM * HDIM);
  cvt_bf16<<<512,  256, 0, stream>>>(Wk, Wkb, HDIM * HDIM);
  cvt_bf16<<<512,  256, 0, stream>>>(Wv, Wvb, HDIM * HDIM);
  cvt_bf16<<<512,  256, 0, stream>>>(Wo, Wob, HDIM * HDIM);

  dim3 ggrid(HDIM / 128, S_LEN / 128);  // (8, 32)
  gemm128<0><<<ggrid, 256, 0, stream>>>(Xb, Wqb, Qb, 0.125f);  // Q pre-scaled
  gemm128<0><<<ggrid, 256, 0, stream>>>(Xb, Wkb, Kb, 1.0f);
  gemm128<1><<<ggrid, 256, 0, stream>>>(Xb, Wvb, Vtb, 1.0f);   // V transposed

  attn_sparse<<<dim3(S_LEN / 16, NHEADS), 32, 0, stream>>>(Qb, Kb, Vtb, Ob);

  gemm128<2><<<ggrid, 256, 0, stream>>>(Ob, Wob, (float*)d_out, 1.0f);
}
